// TorchAttention_49950469652794
// MI455X (gfx1250) — matrix operand order repaired and checked by reading, not compiled, not run on hardware
//
#include <hip/hip_runtime.h>

// ---------------------------------------------------------------------------
// MI455X (gfx1250) attention pipeline.
// wave32 + v_wmma_f32_16x16x32_f16 + global_load_async_to_lds_b128.
// ---------------------------------------------------------------------------

typedef __attribute__((ext_vector_type(16))) _Float16 v16h;
typedef __attribute__((ext_vector_type(8)))  float    v8f;

#define WMMA16(a, b, c) \
  __builtin_amdgcn_wmma_f32_16x16x32_f16(false, (a), false, (b), (short)0, (c), false, false)

// ---- CDNA5 async global->LDS copy (ISA 15.18.3, ASYNCcnt-tracked) ----------
__device__ __forceinline__ uint32_t lds_addr_of(const void* p) {
  // generic LDS pointer: low 32 bits are the LDS byte address (aperture map)
  return (uint32_t)(uintptr_t)p;
}
__device__ __forceinline__ void async_g2l_b128(uint32_t lds, const void* g) {
  asm volatile("global_load_async_to_lds_b128 %0, %1, off"
               :: "v"(lds), "v"(g) : "memory");
}
__device__ __forceinline__ void wait_async0() {
  asm volatile("s_wait_asynccnt 0" ::: "memory");
}

// ---- WMMA fragment loaders (wave32), contiguous k = 16*hf + s in A and B -------------------

// A-matrix 16x32 f16, row-major source, leading dim ld.
// Per lane: one contiguous 32B run -> 2x ds_load_b128.
__device__ __forceinline__ v16h load_a_16x32(const _Float16* base, int ld, int lane) {
  const int r = lane & 15, hf = lane >> 4;
  const _Float16* row = base + r * ld;
  v16h a;
#pragma unroll
  for (int i = 0; i < 8; ++i) {
    int k0 = 2 * i + 16 * hf;
    a[2 * i]     = row[k0];
    a[2 * i + 1] = row[k0 + 1];
  }
  return a;
}

// B-matrix 32x16 built from X[n][k] row-major (transposed operand in LDS):
// B[k][n] = X[n][k]. Per lane: one contiguous 32B run -> 2x ds_load_b128.
__device__ __forceinline__ v16h load_bt_16x32(const _Float16* base, int ld, int lane) {
  const int n = lane & 15, hf = lane >> 4;
  const _Float16* row = base + n * ld;
  v16h b;
#pragma unroll
  for (int i = 0; i < 8; ++i) {
    int k0 = 2 * i + 16 * hf;
    b[2 * i]     = row[k0];
    b[2 * i + 1] = row[k0 + 1];
  }
  return b;
}

// ---------------------------------------------------------------------------
// Kernel 1a: vectorized f32 -> f16 convert (activations). 8 elems/thread-step.
// ---------------------------------------------------------------------------
union H8 { float4 f4; _Float16 h[8]; };

__global__ void f32_to_f16_kernel(const float* __restrict__ in,
                                  _Float16* __restrict__ out, int n8) {
  int i = blockIdx.x * blockDim.x + threadIdx.x;
  const int stride = gridDim.x * blockDim.x;
  for (; i < n8; i += stride) {
    const float4* src = reinterpret_cast<const float4*>(in) + (size_t)i * 2;
    const float4 a = src[0], b = src[1];
    H8 r;
    r.h[0] = (_Float16)a.x; r.h[1] = (_Float16)a.y;
    r.h[2] = (_Float16)a.z; r.h[3] = (_Float16)a.w;
    r.h[4] = (_Float16)b.x; r.h[5] = (_Float16)b.y;
    r.h[6] = (_Float16)b.z; r.h[7] = (_Float16)b.w;
    reinterpret_cast<float4*>(out)[i] = r.f4;
  }
}

// ---------------------------------------------------------------------------
// Kernel 1b: f32 [R][C] -> f16 [C][R] tiled transpose (weights).
// ---------------------------------------------------------------------------
__global__ __launch_bounds__(256) void transpose_f32_to_f16_kernel(
    const float* __restrict__ in, _Float16* __restrict__ out, int R, int C) {
  __shared__ float tile[32][33];
  const int c0 = blockIdx.x * 32, r0 = blockIdx.y * 32;
  const int tx = threadIdx.x, ty = threadIdx.y;  // (32, 8)
#pragma unroll
  for (int i = 0; i < 4; ++i)
    tile[ty + i * 8][tx] = in[(size_t)(r0 + ty + i * 8) * C + c0 + tx];
  __syncthreads();
#pragma unroll
  for (int i = 0; i < 4; ++i)
    out[(size_t)(c0 + ty + i * 8) * R + r0 + tx] = (_Float16)tile[tx][ty + i * 8];
}

// ---------------------------------------------------------------------------
// Kernel 2/5: f16 GEMM  C[M][N] = A[M][K] * Bt[N][K]^T (+bias), f32 out.
// Block tile 128(M) x 64(N), BK=64, async double-buffered LDS staging.
// ---------------------------------------------------------------------------
__global__ __launch_bounds__(256) void gemm_f16_kernel(
    const _Float16* __restrict__ A, const _Float16* __restrict__ Bt,
    float* __restrict__ C, const float* __restrict__ bias,
    int M, int N, int K) {
  __shared__ _Float16 As[2][128 * 72];  // [m][k], row stride 144B
  __shared__ _Float16 Bs[2][64 * 72];   // [n][k]

  const int tid = threadIdx.x;
  const int lane = tid & 31;
  const int wave = tid >> 5;
  const int wm = wave & 3;   // 4 M-slabs of 32 rows
  const int wn = wave >> 2;  // 2 N-slabs of 32 cols
  const int m0 = blockIdx.y * 128;
  const int n0 = blockIdx.x * 64;

  v8f acc00 = {}, acc01 = {}, acc10 = {}, acc11 = {};

  const int ar = tid >> 1, ac = (tid & 1) * 32;  // A copy: 32 f16 / thread
  const int br = tid >> 2, bc = (tid & 3) * 16;  // B copy: 16 f16 / thread
  const _Float16* ga = A + (size_t)(m0 + ar) * K + ac;
  const _Float16* gb = Bt + (size_t)(n0 + br) * K + bc;
  const uint32_t laA[2] = {lds_addr_of(&As[0][ar * 72 + ac]),
                           lds_addr_of(&As[1][ar * 72 + ac])};
  const uint32_t laB[2] = {lds_addr_of(&Bs[0][br * 72 + bc]),
                           lds_addr_of(&Bs[1][br * 72 + bc])};

  // prologue: async-stage tile 0 into buffer 0
#pragma unroll
  for (int c = 0; c < 4; ++c) async_g2l_b128(laA[0] + 16 * c, ga + 8 * c);
#pragma unroll
  for (int c = 0; c < 2; ++c) async_g2l_b128(laB[0] + 16 * c, gb + 8 * c);
  wait_async0();
  __syncthreads();

  const int nt = K >> 6;
  for (int t = 0; t < nt; ++t) {
    const int buf = t & 1;
    if (t + 1 < nt) {  // other buffer's readers passed the previous barrier
      const _Float16* gan = ga + (size_t)(t + 1) * 64;
      const _Float16* gbn = gb + (size_t)(t + 1) * 64;
#pragma unroll
      for (int c = 0; c < 4; ++c) async_g2l_b128(laA[buf ^ 1] + 16 * c, gan + 8 * c);
#pragma unroll
      for (int c = 0; c < 2; ++c) async_g2l_b128(laB[buf ^ 1] + 16 * c, gbn + 8 * c);
    }
    const _Float16* as = As[buf];
    const _Float16* bs = Bs[buf];
#pragma unroll
    for (int kk = 0; kk < 2; ++kk) {
      v16h a0 = load_a_16x32(&as[(32 * wm) * 72 + kk * 32], 72, lane);
      v16h a1 = load_a_16x32(&as[(32 * wm + 16) * 72 + kk * 32], 72, lane);
      v16h b0 = load_bt_16x32(&bs[(32 * wn) * 72 + kk * 32], 72, lane);
      v16h b1 = load_bt_16x32(&bs[(32 * wn + 16) * 72 + kk * 32], 72, lane);
      acc00 = WMMA16(a0, b0, acc00);
      acc01 = WMMA16(a0, b1, acc01);
      acc10 = WMMA16(a1, b0, acc10);
      acc11 = WMMA16(a1, b1, acc11);
    }
    wait_async0();   // next tile landed (overlapped with the WMMAs above)
    __syncthreads();
  }

  // epilogue: C/D layout -> lane 0-15: M=0..7 (vgpr j), lane 16-31: M=8..15
  const int hf = lane >> 4, nn = lane & 15;
  const v8f* accs[2][2] = {{&acc00, &acc01}, {&acc10, &acc11}};
#pragma unroll
  for (int i = 0; i < 2; ++i) {
#pragma unroll
    for (int j = 0; j < 2; ++j) {
      const int gm = m0 + 32 * wm + 16 * i + 8 * hf;
      const int gn = n0 + 32 * wn + 16 * j + nn;
      const float bv = bias ? bias[gn] : 0.0f;
      const v8f a = *accs[i][j];
#pragma unroll
      for (int jj = 0; jj < 8; ++jj)
        C[(size_t)(gm + jj) * N + gn] = a[jj] + bv;
    }
  }
}

// ---------------------------------------------------------------------------
// Kernel 3: LayerNorm(q,k) + q*D^-0.5, pack to f16.
//   Q, K -> [BH][2048 key][64 d]  (row-major)
//   V    -> [BH][64 d][2048 key]  (transposed via LDS tile, coalesced writes)
// grid = (128 token-tiles of 64, 48 = qkvi*16+h), 256 threads.
// ---------------------------------------------------------------------------
__global__ __launch_bounds__(256) void ln_pack_kernel(
    const float* __restrict__ qkv,
    const float* __restrict__ qn_s, const float* __restrict__ qn_b,
    const float* __restrict__ kn_s, const float* __restrict__ kn_b,
    _Float16* __restrict__ Q, _Float16* __restrict__ Kh, _Float16* __restrict__ Vt) {
  __shared__ _Float16 vtile[64 * 72];
  const int lane = threadIdx.x & 31, wave = threadIdx.x >> 5;
  const int tok0 = blockIdx.x * 64;
  const int g = blockIdx.y;
  const int qkvi = g >> 4, h = g & 15;
  const int bh = (tok0 >> 11) * 16 + h;  // constant per 64-token tile
  const int d0 = lane, d1 = lane + 32;

  if (qkvi == 2) {  // V: convert + transpose
#pragma unroll
    for (int it = 0; it < 8; ++it) {
      const int row = wave * 8 + it;
      const float* src = qkv + (size_t)(tok0 + row) * 3072 + 2048 + h * 64;
      vtile[d0 * 72 + row] = (_Float16)src[d0];
      vtile[d1 * 72 + row] = (_Float16)src[d1];
    }
    __syncthreads();
    const int r = threadIdx.x >> 2, c0 = (threadIdx.x & 3) * 16;
    float4* dst = reinterpret_cast<float4*>(
        Vt + ((size_t)bh * 64 + r) * 2048 + (tok0 & 2047) + c0);
    const float4* s = reinterpret_cast<const float4*>(&vtile[r * 72 + c0]);
    dst[0] = s[0]; dst[1] = s[1];
    return;
  }

  const float* sc = (qkvi == 0) ? qn_s : kn_s;
  const float* bi = (qkvi == 0) ? qn_b : kn_b;
  const float mul = (qkvi == 0) ? 0.125f : 1.0f;  // D^-0.5 = 1/8 on q
  _Float16* outbase = (qkvi == 0) ? Q : Kh;
  const float s0 = sc[d0], s1 = sc[d1], b0 = bi[d0], b1 = bi[d1];
#pragma unroll
  for (int it = 0; it < 8; ++it) {
    const int tok = tok0 + wave * 8 + it;
    const float* src = qkv + (size_t)tok * 3072 + qkvi * 1024 + h * 64;
    const float x0 = src[d0], x1 = src[d1];
    float s = x0 + x1;
#pragma unroll
    for (int m = 1; m < 32; m <<= 1) s += __shfl_xor(s, m, 32);
    const float mu = s * (1.0f / 64.0f);
    float vv = (x0 - mu) * (x0 - mu) + (x1 - mu) * (x1 - mu);
#pragma unroll
    for (int m = 1; m < 32; m <<= 1) vv += __shfl_xor(vv, m, 32);
    const float r = rsqrtf(vv * (1.0f / 64.0f) + 1e-5f);
    _Float16* dst = outbase + ((size_t)bh * 2048 + (tok & 2047)) * 64;
    dst[d0] = (_Float16)(((x0 - mu) * r * s0 + b0) * mul);
    dst[d1] = (_Float16)(((x1 - mu) * r * s1 + b1) * mul);
  }
}

// ---------------------------------------------------------------------------
// Kernel 4: flash attention. grid = (N/128, BH). 8 waves; wave owns 16 q rows.
// KV streamed in 64-key tiles via async LDS loads. V pre-transposed [d][key].
// ---------------------------------------------------------------------------
__global__ __launch_bounds__(256) void flash_attn_kernel(
    const _Float16* __restrict__ Q, const _Float16* __restrict__ Kg,
    const _Float16* __restrict__ Vt, _Float16* __restrict__ O) {
  __shared__ _Float16 Qs[128 * 72];      // [q row][d]
  __shared__ _Float16 Ks[64 * 72];       // [key][d]
  __shared__ _Float16 Vs[64 * 72];       // [d][key]
  __shared__ _Float16 Ps[8 * 16 * 72];   // per-wave P staging [m][key]

  const int tid = threadIdx.x, lane = tid & 31, wave = tid >> 5;
  const int bh = blockIdx.y;
  const int q0 = blockIdx.x * 128;
  const size_t baseqk = (size_t)bh * 2048 * 64;
  const size_t basev  = (size_t)bh * 64 * 2048;

  // Q tile 128x64 -> LDS (async)
  {
    const int r = tid >> 1, cb = (tid & 1) * 32;
    const _Float16* g = Q + baseqk + (size_t)(q0 + r) * 64 + cb;
    const uint32_t l = lds_addr_of(&Qs[r * 72 + cb]);
#pragma unroll
    for (int c = 0; c < 4; ++c) async_g2l_b128(l + 16 * c, g + 8 * c);
  }
  wait_async0();
  __syncthreads();
  const v16h aq0 = load_a_16x32(&Qs[(16 * wave) * 72 + 0], 72, lane);
  const v16h aq1 = load_a_16x32(&Qs[(16 * wave) * 72 + 32], 72, lane);

  v8f o0 = {}, o1 = {}, o2 = {}, o3 = {};
  float mrow[8], lrow[8];
#pragma unroll
  for (int j = 0; j < 8; ++j) { mrow[j] = -3.0e38f; lrow[j] = 0.0f; }

  const int hf = lane >> 4, nn = lane & 15;
  const int kr = tid >> 2, kc = (tid & 3) * 16;
  const uint32_t lK = lds_addr_of(&Ks[kr * 72 + kc]);
  const uint32_t lV = lds_addr_of(&Vs[kr * 72 + kc]);
  _Float16* pw = &Ps[wave * 16 * 72];

  for (int kt = 0; kt < 2048; kt += 64) {
    __syncthreads();  // readers of the previous tile are done
    {
      const _Float16* gk = Kg + baseqk + (size_t)(kt + kr) * 64 + kc;
      const _Float16* gv = Vt + basev + (size_t)kr * 2048 + kt + kc;
      async_g2l_b128(lK, gk);      async_g2l_b128(lK + 16, gk + 8);
      async_g2l_b128(lV, gv);      async_g2l_b128(lV + 16, gv + 8);
    }
    if (kt + 64 < 2048) {  // warm L2 for the next tile
      __builtin_prefetch(Kg + baseqk + (size_t)(kt + 64 + kr) * 64 + kc, 0, 0);
      __builtin_prefetch(Vt + basev + (size_t)kr * 2048 + kt + 64 + kc, 0, 0);
    }
    wait_async0();
    __syncthreads();

    // S = Q K^T  (16 x 64 per wave; B[k][n] = K[n-th key][k] -> bt frags)
    v8f s4[4];
#pragma unroll
    for (int jn = 0; jn < 4; ++jn) {
      v8f s = {};
      v16h b0 = load_bt_16x32(&Ks[(jn * 16) * 72 + 0], 72, lane);
      s = WMMA16(aq0, b0, s);
      v16h b1 = load_bt_16x32(&Ks[(jn * 16) * 72 + 32], 72, lane);
      s = WMMA16(aq1, b1, s);
      s4[jn] = s;
    }

    // online softmax: row = 8*hf + j, spread over 16 lanes (n) per half
    float nm[8];
#pragma unroll
    for (int j = 0; j < 8; ++j) {
      float m = fmaxf(fmaxf(s4[0][j], s4[1][j]), fmaxf(s4[2][j], s4[3][j]));
      nm[j] = fmaxf(mrow[j], m);
    }
#pragma unroll
    for (int j = 0; j < 8; ++j) {
#pragma unroll
      for (int msk = 1; msk < 16; msk <<= 1)
        nm[j] = fmaxf(nm[j], __shfl_xor(nm[j], msk, 32));
    }
    float alpha[8], rs[8];
#pragma unroll
    for (int j = 0; j < 8; ++j) {
      alpha[j] = __expf(mrow[j] - nm[j]);
      mrow[j] = nm[j];
      rs[j] = 0.0f;
    }
#pragma unroll
    for (int jn = 0; jn < 4; ++jn) {
#pragma unroll
      for (int j = 0; j < 8; ++j) {
        float p = __expf(s4[jn][j] - nm[j]);
        s4[jn][j] = p;
        rs[j] += p;
      }
    }
#pragma unroll
    for (int j = 0; j < 8; ++j) {
#pragma unroll
      for (int msk = 1; msk < 16; msk <<= 1) rs[j] += __shfl_xor(rs[j], msk, 32);
      lrow[j] = lrow[j] * alpha[j] + rs[j];
    }
#pragma unroll
    for (int j = 0; j < 8; ++j) {
      o0[j] *= alpha[j]; o1[j] *= alpha[j]; o2[j] *= alpha[j]; o3[j] *= alpha[j];
    }

    // P (f32 C-layout) -> per-wave LDS slab as f16 [m][key]
#pragma unroll
    for (int jn = 0; jn < 4; ++jn) {
#pragma unroll
      for (int j = 0; j < 8; ++j)
        pw[(8 * hf + j) * 72 + jn * 16 + nn] = (_Float16)s4[jn][j];
    }

    // O += P V : A = P[m][key], B[k][n] = V[key k][d n] = Vs[n][k] -> bt frags
    v16h ap0 = load_a_16x32(pw + 0, 72, lane);
    v16h ap1 = load_a_16x32(pw + 32, 72, lane);
    {
      v16h b0 = load_bt_16x32(&Vs[(0 * 16) * 72 + 0], 72, lane);
      o0 = WMMA16(ap0, b0, o0);
      v16h b1 = load_bt_16x32(&Vs[(0 * 16) * 72 + 32], 72, lane);
      o0 = WMMA16(ap1, b1, o0);
    }
    {
      v16h b0 = load_bt_16x32(&Vs[(1 * 16) * 72 + 0], 72, lane);
      o1 = WMMA16(ap0, b0, o1);
      v16h b1 = load_bt_16x32(&Vs[(1 * 16) * 72 + 32], 72, lane);
      o1 = WMMA16(ap1, b1, o1);
    }
    {
      v16h b0 = load_bt_16x32(&Vs[(2 * 16) * 72 + 0], 72, lane);
      o2 = WMMA16(ap0, b0, o2);
      v16h b1 = load_bt_16x32(&Vs[(2 * 16) * 72 + 32], 72, lane);
      o2 = WMMA16(ap1, b1, o2);
    }
    {
      v16h b0 = load_bt_16x32(&Vs[(3 * 16) * 72 + 0], 72, lane);
      o3 = WMMA16(ap0, b0, o3);
      v16h b1 = load_bt_16x32(&Vs[(3 * 16) * 72 + 32], 72, lane);
      o3 = WMMA16(ap1, b1, o3);
    }
  }

  // normalize + write token-major f16 [B][N][C] for the proj GEMM
  const int b = bh >> 4, h = bh & 15;
  _Float16* outp = O + ((size_t)b * 2048 + q0 + 16 * wave) * 1024 + h * 64;
  const v8f* os[4] = {&o0, &o1, &o2, &o3};
#pragma unroll
  for (int j = 0; j < 8; ++j) {
    const int row = 8 * hf + j;
    const float inv = 1.0f / lrow[j];
#pragma unroll
    for (int jn = 0; jn < 4; ++jn)
      outp[(size_t)row * 1024 + jn * 16 + nn] = (_Float16)((*os[jn])[j] * inv);
  }
}

// ---------------------------------------------------------------------------
// Host-side orchestration
// ---------------------------------------------------------------------------
extern "C" void kernel_launch(void* const* d_in, const int* in_sizes, int n_in,
                              void* d_out, int out_size, void* d_ws, size_t ws_size,
                              hipStream_t stream) {
  (void)in_sizes; (void)n_in; (void)out_size; (void)ws_size;

  const float* x      = (const float*)d_in[0];  // [4,2048,1024]
  const float* qkv_w  = (const float*)d_in[1];  // [1024,3072]
  const float* proj_w = (const float*)d_in[2];  // [1024,1024]
  const float* proj_b = (const float*)d_in[3];  // [1024]
  const float* qn_s   = (const float*)d_in[4];
  const float* qn_b   = (const float*)d_in[5];
  const float* kn_s   = (const float*)d_in[6];
  const float* kn_b   = (const float*)d_in[7];
  float* out = (float*)d_out;

  char* ws = (char*)d_ws;
  _Float16* x_h     = (_Float16*)ws; ws += (size_t)8192 * 1024 * 2;    // 16 MB
  _Float16* wqkv_t  = (_Float16*)ws; ws += (size_t)3072 * 1024 * 2;    //  6 MB [3072][1024]
  _Float16* wproj_t = (_Float16*)ws; ws += (size_t)1024 * 1024 * 2;    //  2 MB [1024][1024]
  float*    qkv_f   = (float*)ws;    ws += (size_t)8192 * 3072 * 4;    // 96 MB
  _Float16* Qh      = (_Float16*)ws; ws += (size_t)64 * 2048 * 64 * 2; // 16 MB [BH][N][D]
  _Float16* Kh      = (_Float16*)ws; ws += (size_t)64 * 2048 * 64 * 2; // 16 MB [BH][N][D]
  _Float16* Vth     = (_Float16*)ws; ws += (size_t)64 * 64 * 2048 * 2; // 16 MB [BH][D][N]
  _Float16* Ah      = (_Float16*)ws; ws += (size_t)8192 * 1024 * 2;    // 16 MB

  // 1) activations -> f16; weights -> f16 transposed ([N][K] for bt-frags)
  f32_to_f16_kernel<<<2048, 256, 0, stream>>>(x, x_h, (8192 * 1024) / 8);
  transpose_f32_to_f16_kernel<<<dim3(96, 32), dim3(32, 8), 0, stream>>>(
      qkv_w, wqkv_t, 1024, 3072);
  transpose_f32_to_f16_kernel<<<dim3(32, 32), dim3(32, 8), 0, stream>>>(
      proj_w, wproj_t, 1024, 1024);

  // 2) QKV GEMM: [8192,1024] x [1024,3072] -> f32
  gemm_f16_kernel<<<dim3(48, 64), 256, 0, stream>>>(x_h, wqkv_t, qkv_f, nullptr,
                                                    8192, 3072, 1024);

  // 3) LayerNorm(q,k) + scale; pack Q/K row-major, V transposed
  ln_pack_kernel<<<dim3(128, 48), 256, 0, stream>>>(qkv_f, qn_s, qn_b, kn_s, kn_b,
                                                    Qh, Kh, Vth);

  // 4) flash attention -> token-major f16 [8192,1024]
  flash_attn_kernel<<<dim3(16, 64), 256, 0, stream>>>(Qh, Kh, Vth, Ah);

  // 5) proj GEMM + bias -> f32 out
  gemm_f16_kernel<<<dim3(16, 64), 256, 0, stream>>>(Ah, wproj_t, out, proj_b,
                                                    8192, 1024, 1024);
}
